// CausalTransformerShard_77318001262956
// MI455X (gfx1250) — matrix operand order repaired and checked by reading, not compiled, not run on hardware
//
#include <hip/hip_runtime.h>
#include <hip/hip_bf16.h>

#define DIM 2048
#define HEADS 16
#define LAYERS 4
#define SEQ 1024
#define VOCAB 32000
#define DPH 128
#define FFN 8192
#define NBUCK 32

typedef __bf16 bf16_t;
typedef __attribute__((ext_vector_type(16))) __bf16 v16bf;
typedef __attribute__((ext_vector_type(8)))  __bf16 v8bf;
typedef __attribute__((ext_vector_type(2)))  __bf16 v2bf;
typedef __attribute__((ext_vector_type(8)))  float  v8f;

// Pack two floats -> packed bf16 pair using native converts (RNE).
__device__ __forceinline__ unsigned pk2bf(float a, float b) {
    v2bf v;
    v.x = (__bf16)a;
    v.y = (__bf16)b;
    return __builtin_bit_cast(unsigned, v);
}

__device__ __forceinline__ float gelu_tanh(float x) {
    float x3 = x * x * x;
    return 0.5f * x * (1.0f + tanhf(0.7978845608028654f * (x + 0.044715f * x3)));
}

// ---------------------------------------------------------------------------
// Embedding gather: x[t][d] = w_embed[ctx[t]][d] + b_embed[d]
// ---------------------------------------------------------------------------
__global__ __launch_bounds__(256) void k_embed(const int* __restrict__ ctx,
                                               const float* __restrict__ we,
                                               const float* __restrict__ be,
                                               float* __restrict__ x) {
    int t = blockIdx.x;
    int c = ctx[t];
    const float* src = we + (long long)c * DIM;
    float* dst = x + (long long)t * DIM;
    for (int d = threadIdx.x; d < DIM; d += 256)
        dst[d] = src[d] + be[d];
}

// ---------------------------------------------------------------------------
// LayerNorm, one block per row
// ---------------------------------------------------------------------------
__global__ __launch_bounds__(256) void k_layernorm(const float* __restrict__ x,
                                                   const float* __restrict__ scale,
                                                   const float* __restrict__ offset,
                                                   float* __restrict__ h) {
    __shared__ float s1[256], s2[256];
    int t = blockIdx.x, tid = threadIdx.x;
    const float* xr = x + (long long)t * DIM;
    float a = 0.f, b = 0.f;
    for (int d = tid; d < DIM; d += 256) { float v = xr[d]; a += v; b += v * v; }
    s1[tid] = a; s2[tid] = b;
    __syncthreads();
    for (int s = 128; s > 0; s >>= 1) {
        if (tid < s) { s1[tid] += s1[tid + s]; s2[tid] += s2[tid + s]; }
        __syncthreads();
    }
    float mean = s1[0] * (1.0f / DIM);
    float var  = s2[0] * (1.0f / DIM) - mean * mean;
    float rstd = rsqrtf(var + 1e-5f);
    float* hr = h + (long long)t * DIM;
    for (int d = tid; d < DIM; d += 256)
        hr[d] = scale[d] * (xr[d] - mean) * rstd + offset[d];
}

// ---------------------------------------------------------------------------
// Generic batched WMMA GEMM:  C = alpha * A(MxK) * B  [+bias][gelu][+resid]
// A fp32 row-major (lda). B fp32: BT ? NxK row-major (ldb) : KxN (ldb).
// C fp32 row-major (ldc). fp32->bf16 conversion happens in the LDS fill.
// Block 256 threads (8 waves), tile 128x128, K-step 32, double-buffered LDS.
// Wave w: rows (w&3)*32..+31, cols (w>>2)*64..+63 -> 2x4 wmma tiles.
// ---------------------------------------------------------------------------
#define BM 128
#define BN 128
#define BKK 32
#define SK 40   // padded LDS K-stride (bf16 elems); 80B rows keep 16B alignment

template <bool BT, bool GELU>
__global__ __launch_bounds__(256) void k_gemm_wmma(
    const float* __restrict__ A, int lda, long long sA,
    const float* __restrict__ B, int ldb, long long sB,
    float* __restrict__ C, int ldc, long long sC,
    const float* __restrict__ bias,
    const float* __restrict__ resid,
    int M, int N, int K, float alpha) {

    __shared__ bf16_t As[2][BM * SK];
    __shared__ bf16_t Bs[2][BN * SK];

    int bz = blockIdx.z;
    A += (long long)bz * sA;
    B += (long long)bz * sB;
    C += (long long)bz * sC;

    int tid  = threadIdx.x;
    int lane = tid & 31;
    int wave = tid >> 5;
    int g    = lane >> 4;     // half-wave group 0/1
    int l16  = lane & 15;
    int wm   = (wave & 3) * 32;
    int wn   = (wave >> 2) * 64;
    int row0 = blockIdx.y * BM;
    int col0 = blockIdx.x * BN;

    // Staging coordinates (constant per thread)
    const int a_r  = tid >> 3;             // 0..31 (+32 per it)
    const int a_kq = (tid & 7) << 2;       // 0,4,..28
    const int bt_n = tid >> 3;
    const int bn_k = tid >> 5;             // 0..7 (+8 per it)
    const int bn_q = (tid & 31) << 2;      // 0..124

    float4 areg[4];
    float4 breg[4];

    auto load_tiles = [&](int kb) {
#pragma unroll
        for (int it = 0; it < 4; ++it)
            areg[it] = *(const float4*)(A + (long long)(row0 + a_r + it * 32) * lda + kb + a_kq);
        if (BT) {
#pragma unroll
            for (int it = 0; it < 4; ++it)
                breg[it] = *(const float4*)(B + (long long)(col0 + bt_n + it * 32) * ldb + kb + a_kq);
        } else {
#pragma unroll
            for (int it = 0; it < 4; ++it)
                breg[it] = *(const float4*)(B + (long long)(kb + bn_k + it * 8) * ldb + col0 + bn_q);
        }
    };

    auto store_tiles = [&](int buf) {
        bf16_t* as = As[buf];
        bf16_t* bs = Bs[buf];
#pragma unroll
        for (int it = 0; it < 4; ++it) {
            uint2 p;
            p.x = pk2bf(areg[it].x, areg[it].y);
            p.y = pk2bf(areg[it].z, areg[it].w);
            *(uint2*)(&as[(a_r + it * 32) * SK + a_kq]) = p;
        }
        if (BT) {
#pragma unroll
            for (int it = 0; it < 4; ++it) {
                uint2 p;
                p.x = pk2bf(breg[it].x, breg[it].y);
                p.y = pk2bf(breg[it].z, breg[it].w);
                *(uint2*)(&bs[(bt_n + it * 32) * SK + a_kq]) = p;
            }
        } else {
#pragma unroll
            for (int it = 0; it < 4; ++it) {
                int k = bn_k + it * 8;
                bs[(bn_q + 0) * SK + k] = (__bf16)breg[it].x;
                bs[(bn_q + 1) * SK + k] = (__bf16)breg[it].y;
                bs[(bn_q + 2) * SK + k] = (__bf16)breg[it].z;
                bs[(bn_q + 3) * SK + k] = (__bf16)breg[it].w;
            }
        }
    };

    v8f acc[2][4];
#pragma unroll
    for (int i = 0; i < 2; ++i)
#pragma unroll
        for (int j = 0; j < 4; ++j) {
            v8f z = {0.f, 0.f, 0.f, 0.f, 0.f, 0.f, 0.f, 0.f};
            acc[i][j] = z;
        }

    // Prologue: stage tile 0
    load_tiles(0);
    store_tiles(0);
    __syncthreads();

    const int nk = K / BKK;
    for (int i = 0; i < nk; ++i) {
        const bool has_next = (i + 1) < nk;
        // Issue next tile's global loads; latency hidden under the WMMAs.
        if (has_next) load_tiles((i + 1) * BKK);

        const bf16_t* as = As[i & 1];
        const bf16_t* bs = Bs[i & 1];

        v16bf afrag[2];
#pragma unroll
        for (int tm = 0; tm < 2; ++tm) {
            int mr = wm + tm * 16 + l16;
            v8bf alo = *(const v8bf*)(&as[mr * SK + g * 8]);
            v8bf ahi = *(const v8bf*)(&as[mr * SK + 16 + g * 8]);
            afrag[tm] = __builtin_shufflevector(alo, ahi,
                0, 1, 2, 3, 4, 5, 6, 7, 8, 9, 10, 11, 12, 13, 14, 15);
        }
#pragma unroll
        for (int tn = 0; tn < 4; ++tn) {
            int nr = wn + tn * 16 + l16;
            v8bf blo = *(const v8bf*)(&bs[nr * SK + g * 8]);
            v8bf bhi = *(const v8bf*)(&bs[nr * SK + 16 + g * 8]);
            v16bf bfrag = __builtin_shufflevector(blo, bhi,
                0, 1, 2, 3, 4, 5, 6, 7, 8, 9, 10, 11, 12, 13, 14, 15);
#pragma unroll
            for (int tm = 0; tm < 2; ++tm) {
                acc[tm][tn] = __builtin_amdgcn_wmma_f32_16x16x32_bf16(
                    false, afrag[tm], false, bfrag, (short)0, acc[tm][tn], false, false);
            }
        }

        if (has_next) {
            store_tiles((i + 1) & 1);   // other buffer: safe to fill while peers compute
            __syncthreads();
        }
    }

    // ---- epilogue: alpha, bias, gelu, residual ----
#pragma unroll
    for (int tm = 0; tm < 2; ++tm)
#pragma unroll
        for (int tn = 0; tn < 4; ++tn) {
            int c = col0 + wn + tn * 16 + l16;
            float bv = bias ? bias[c] : 0.f;
#pragma unroll
            for (int e = 0; e < 8; ++e) {
                int r = row0 + wm + tm * 16 + g * 8 + e;
                float v = acc[tm][tn][e] * alpha + bv;
                if (GELU) v = gelu_tanh(v);
                if (resid) v += resid[(long long)r * ldc + c];
                C[(long long)r * ldc + c] = v;
            }
        }
}

// ---------------------------------------------------------------------------
// Causal softmax with T5 relative-position bias. One block per (row i, head).
// scores laid out [head][i][j], scores already scaled by 1/sqrt(DPH).
// ---------------------------------------------------------------------------
__global__ __launch_bounds__(256) void k_softmax_bias(float* __restrict__ scores,
                                                      const float* __restrict__ rel) {
    __shared__ float red[256];
    int i  = blockIdx.x;
    int hh = blockIdx.y;
    int tid = threadIdx.x;
    float* row = scores + ((long long)hh * SEQ + i) * SEQ;

    float vals[4];
    float mx = -3.0e38f;
#pragma unroll
    for (int c = 0; c < 4; ++c) {
        int j = tid + c * 256;
        float v;
        if (j > i) {
            v = -1e10f;
        } else {
            int n = i - j;   // n = -(j - i) clamped >= 0
            int bucket;
            if (n < 16) {
                bucket = n;
            } else {
                int vb = 16 + (int)(logf((float)n * (1.0f / 16.0f) + 1.1920929e-7f)
                                    * (16.0f / logf(8.0f)));
                bucket = vb < 31 ? vb : 31;
            }
            v = row[j] + rel[hh * NBUCK + bucket];
        }
        vals[c] = v;
        mx = fmaxf(mx, v);
    }
    red[tid] = mx; __syncthreads();
    for (int s = 128; s > 0; s >>= 1) {
        if (tid < s) red[tid] = fmaxf(red[tid], red[tid + s]);
        __syncthreads();
    }
    mx = red[0]; __syncthreads();

    float sum = 0.f;
#pragma unroll
    for (int c = 0; c < 4; ++c) {
        vals[c] = expf(vals[c] - mx);
        sum += vals[c];
    }
    red[tid] = sum; __syncthreads();
    for (int s = 128; s > 0; s >>= 1) {
        if (tid < s) red[tid] += red[tid + s];
        __syncthreads();
    }
    float inv = 1.0f / red[0];
#pragma unroll
    for (int c = 0; c < 4; ++c)
        row[tid + c * 256] = vals[c] * inv;
}

// ---------------------------------------------------------------------------
// Loss: per row, loss[t] = logsumexp(logits[t]) - logits[t][target[t]]
// ---------------------------------------------------------------------------
__global__ __launch_bounds__(256) void k_loss(const float* __restrict__ logits,
                                              const int* __restrict__ target,
                                              float* __restrict__ loss) {
    __shared__ float red[256];
    int t = blockIdx.x, tid = threadIdx.x;
    const float* lr = logits + (long long)t * VOCAB;

    float mx = -3.0e38f;
    for (int j = tid; j < VOCAB; j += 256) mx = fmaxf(mx, lr[j]);
    red[tid] = mx; __syncthreads();
    for (int s = 128; s > 0; s >>= 1) {
        if (tid < s) red[tid] = fmaxf(red[tid], red[tid + s]);
        __syncthreads();
    }
    mx = red[0]; __syncthreads();

    float sum = 0.f;
    for (int j = tid; j < VOCAB; j += 256) sum += expf(lr[j] - mx);
    red[tid] = sum; __syncthreads();
    for (int s = 128; s > 0; s >>= 1) {
        if (tid < s) red[tid] += red[tid + s];
        __syncthreads();
    }
    if (tid == 0)
        loss[t] = logf(red[0]) + mx - lr[target[t]];
}

// ---------------------------------------------------------------------------
extern "C" void kernel_launch(void* const* d_in, const int* in_sizes, int n_in,
                              void* d_out, int out_size, void* d_ws, size_t ws_size,
                              hipStream_t stream) {
    const int*   ctx      = (const int*)d_in[0];
    const int*   tgt      = (const int*)d_in[1];
    const float* w_embed  = (const float*)d_in[2];
    const float* b_embed  = (const float*)d_in[3];
    const float* rel      = (const float*)d_in[4];
    const float* ln_scale = (const float*)d_in[5];
    const float* ln_off   = (const float*)d_in[6];
    const float* wq       = (const float*)d_in[7];
    const float* wk       = (const float*)d_in[8];
    const float* wv       = (const float*)d_in[9];
    const float* wo       = (const float*)d_in[10];
    const float* w1       = (const float*)d_in[11];
    const float* b1       = (const float*)d_in[12];
    const float* w2       = (const float*)d_in[13];
    const float* b2       = (const float*)d_in[14];
    const float* w_out    = (const float*)d_in[15];
    const float* b_out    = (const float*)d_in[16];
    float* loss = (float*)d_out;

    float* ws = (float*)d_ws;
    size_t off = 0;
    float* x   = ws + off; off += (size_t)SEQ * DIM;
    float* h   = ws + off; off += (size_t)SEQ * DIM;
    float* q   = ws + off; off += (size_t)SEQ * DIM;
    float* kx  = ws + off; off += (size_t)SEQ * DIM;
    float* vx  = ws + off; off += (size_t)SEQ * DIM;
    float* sc  = ws + off; off += (size_t)HEADS * SEQ * SEQ;
    float* av  = ws + off; off += (size_t)SEQ * DIM;
    float* t1  = ws + off; off += (size_t)SEQ * DIM;
    float* fm  = ws + off; off += (size_t)SEQ * FFN;
    float* lg  = ws + off; off += (size_t)SEQ * VOCAB;

    const float inv_sqrt_d = 0.08838834764831845f;  // 1/sqrt(128)
    const long long DD = (long long)DIM * DIM;

    k_embed<<<SEQ, 256, 0, stream>>>(ctx, w_embed, b_embed, x);

    for (int l = 0; l < LAYERS; ++l) {
        k_layernorm<<<SEQ, 256, 0, stream>>>(x, ln_scale + l * DIM, ln_off + l * DIM, h);

        dim3 gqkv(DIM / BN, SEQ / BM, 1);
        k_gemm_wmma<false, false><<<gqkv, 256, 0, stream>>>(
            h, DIM, 0, wq + l * DD, DIM, 0, q, DIM, 0,
            nullptr, nullptr, SEQ, DIM, DIM, 1.0f);
        k_gemm_wmma<false, false><<<gqkv, 256, 0, stream>>>(
            h, DIM, 0, wk + l * DD, DIM, 0, kx, DIM, 0,
            nullptr, nullptr, SEQ, DIM, DIM, 1.0f);
        k_gemm_wmma<false, false><<<gqkv, 256, 0, stream>>>(
            h, DIM, 0, wv + l * DD, DIM, 0, vx, DIM, 0,
            nullptr, nullptr, SEQ, DIM, DIM, 1.0f);

        // scores[h] = (Q_h @ K_h^T) / sqrt(d)   (B transposed mode)
        dim3 gqk(SEQ / BN, SEQ / BM, HEADS);
        k_gemm_wmma<true, false><<<gqk, 256, 0, stream>>>(
            q, DIM, DPH, kx, DIM, DPH, sc, SEQ, (long long)SEQ * SEQ,
            nullptr, nullptr, SEQ, SEQ, DPH, inv_sqrt_d);

        dim3 gsm(SEQ, HEADS);
        k_softmax_bias<<<gsm, 256, 0, stream>>>(sc, rel);

        // av[:, h*DPH:...] = probs_h @ V_h
        dim3 gav(DPH / BN, SEQ / BM, HEADS);
        k_gemm_wmma<false, false><<<gav, 256, 0, stream>>>(
            sc, SEQ, (long long)SEQ * SEQ, vx, DIM, DPH, av, DIM, DPH,
            nullptr, nullptr, SEQ, DPH, SEQ, 1.0f);

        // t1 = x + av @ Wo
        dim3 go(DIM / BN, SEQ / BM, 1);
        k_gemm_wmma<false, false><<<go, 256, 0, stream>>>(
            av, DIM, 0, wo + l * DD, DIM, 0, t1, DIM, 0,
            nullptr, x, SEQ, DIM, DIM, 1.0f);

        // fm = gelu(h @ W1 + b1)
        dim3 g1(FFN / BN, SEQ / BM, 1);
        k_gemm_wmma<false, true><<<g1, 256, 0, stream>>>(
            h, DIM, 0, w1 + (long long)l * DIM * FFN, FFN, 0, fm, FFN, 0,
            b1 + (long long)l * FFN, nullptr, SEQ, FFN, DIM, 1.0f);

        // x = t1 + fm @ W2 + b2
        dim3 g2(DIM / BN, SEQ / BM, 1);
        k_gemm_wmma<false, false><<<g2, 256, 0, stream>>>(
            fm, FFN, 0, w2 + (long long)l * FFN * DIM, DIM, 0, x, DIM, 0,
            b2 + (long long)l * DIM, t1, SEQ, DIM, FFN, 1.0f);
    }

    dim3 gout(VOCAB / BN, SEQ / BM, 1);
    k_gemm_wmma<false, false><<<gout, 256, 0, stream>>>(
        x, DIM, 0, w_out, VOCAB, 0, lg, VOCAB, 0,
        b_out, nullptr, SEQ, VOCAB, DIM, 1.0f);

    k_loss<<<SEQ, 256, 0, stream>>>(lg, tgt, loss);
}
